// ScaledDotProductAttention_30580167147801
// MI455X (gfx1250) — matrix operand order repaired and checked by reading, not compiled, not run on hardware
//
#include <hip/hip_runtime.h>

// ---------------------------------------------------------------------------
// Scaled dot-product attention for MI455X (gfx1250), wave32 + WMMA f16/f32.
//
// Shapes: B=2, H=16, N=2048, D=64.  Outputs: O (B,H,N,D) then attn (B,H,N,N),
// concatenated flat in d_out (reference returns a tuple).
//
// Strategy: per (b,h, 16-row q-tile) block (grid 128 x 32, 128 thr = 4 waves):
//   1) S = (Q/8) K^T via v_wmma_f32_16x16x32_f16, masked, into a 16x2048 f32
//      LDS strip (128 KB; CDNA5 has 320 KB LDS/WGP -> 2 blocks resident).
//   2) per-row max/sum via wave32 shfl_xor reductions.
//   3) normalize, stream P to the attn output (the 536 MB HBM-bound part),
//      keep P in LDS.
//   4) O = P V via v_wmma_f32_16x16x32_f16, f32 accumulate.
// ---------------------------------------------------------------------------

typedef __attribute__((ext_vector_type(16))) _Float16 v16h;
typedef __attribute__((ext_vector_type(8)))  float    v8f;

namespace {
constexpr int kB = 2;
constexpr int kH = 16;
constexpr int kN = 2048;
constexpr int kD = 64;
constexpr float kInvT    = 0.125f;          // 1/TEMPERATURE, exact in fp16
constexpr float kNegInf  = -1000000000.0f;  // reference NEG_INF
constexpr int kSRow = kN + 4;               // padded LDS row stride (bank decorrelation)
}

__global__ __launch_bounds__(128)
void ScaledDotProductAttention_30580167147801_kernel(
    const float* __restrict__ Q, const float* __restrict__ K,
    const float* __restrict__ V, const int* __restrict__ Mask,
    float* __restrict__ OutO, float* __restrict__ OutAttn)
{
    extern __shared__ char smem_raw[];
    float* sS   = (float*)smem_raw;                       // [16][kSRow]
    float* rowM = (float*)(smem_raw + 16 * kSRow * 4);    // [16] row max
    float* rowR = rowM + 16;                              // [16] 1/rowsum
    int*   mrow = (int*)(rowR + 16);                      // [16] row mask bits

    const int tid  = threadIdx.x;
    const int wave = tid >> 5;
    const int lane = tid & 31;
    const int qt   = blockIdx.x;        // q-tile: 0..127
    const int bh   = blockIdx.y;        // fused (b,h): 0..31
    const int b    = bh >> 4;           // H = 16
    const int q0   = qt * 16;

    const float* Qb = Q + (size_t)bh * kN * kD;
    const float* Kb = K + (size_t)bh * kN * kD;
    const float* Vb = V + (size_t)bh * kN * kD;
    const int*   mb = Mask + b * kN;    // mask is (B,1,N)

    if (tid < 16) mrow[tid] = mb[q0 + tid];
    __syncthreads();

    // Fragment coordinate helpers (A/B in this code's contiguous k order, not AMD's table order; C/D f32 layout):
    //  A 16x32: lanes 0-15 -> M=lane, K=0..15; lanes 16-31 -> M=lane-16,
    //           K=16..31. Element a[i] (i<8) = K=koffA+i, a[8+i] = K=8+koffA+i.
    //  B 32x16: lanes 0-15 -> N=lane, K=0..15 (b[i]=K=i); lanes 16-31 -> K=16..31.
    //  C/D 16x16 f32: VGPR j -> M = j + (lane>=16 ? 8 : 0), N = lane&15.
    const int mA    = lane & 15;
    const int koffA = (lane < 16) ? 0 : 16;
    const int nB    = lane & 15;
    const int koffB = (lane < 16) ? 0 : 16;
    const int rbase = (lane < 16) ? 0 : 8;

    // ---- Pass 1: S = (Q/T) K^T, masked, into LDS -------------------------
    // Q A-fragments for d = 0..31 (a0) and d = 32..63 (a1); invariant over k-tiles.
    v16h a0, a1;
    {
        const float* qrow = Qb + (size_t)(q0 + mA) * kD;
        #pragma unroll
        for (int i = 0; i < 8; ++i) {
            a0[i]     = (_Float16)(qrow[ 0 + koffA + i] * kInvT);
            a0[8 + i] = (_Float16)(qrow[ 8 + koffA + i] * kInvT);
            a1[i]     = (_Float16)(qrow[32 + koffA + i] * kInvT);
            a1[8 + i] = (_Float16)(qrow[40 + koffA + i] * kInvT);
        }
    }

    for (int kt = wave; kt < kN / 16; kt += 4) {
        const int kbase = kt * 16;
        const float* krow = Kb + (size_t)(kbase + nB) * kD;   // 16 contiguous f32 per lane
        v16h b0, b1;
        #pragma unroll
        for (int i = 0; i < 16; ++i) {
            b0[i] = (_Float16)krow[ 0 + koffB + i];   // d =  0..31 slice
            b1[i] = (_Float16)krow[32 + koffB + i];   // d = 32..63 slice
        }
        v8f c = {};
        c = __builtin_amdgcn_wmma_f32_16x16x32_f16(false, a0, false, b0,
                                                   (short)0, c, false, false);
        c = __builtin_amdgcn_wmma_f32_16x16x32_f16(false, a1, false, b1,
                                                   (short)0, c, false, false);
        const int mcol = mb[kbase + nB];
        #pragma unroll
        for (int j = 0; j < 8; ++j) {
            const int m = j + rbase;
            const float val = (mcol | mrow[m]) ? kNegInf : c[j];
            sS[m * kSRow + kbase + nB] = val;
        }
    }
    __syncthreads();

    // ---- Pass 2: row max & exp-sum (4 rows per wave, wave32 shfl reduce) --
    for (int r = wave * 4; r < wave * 4 + 4; ++r) {
        const float* Sr = sS + r * kSRow;
        float m = -3.4e38f;
        for (int c = lane; c < kN; c += 32) m = fmaxf(m, Sr[c]);
        #pragma unroll
        for (int off = 16; off > 0; off >>= 1) m = fmaxf(m, __shfl_xor(m, off, 32));
        float s = 0.0f;
        for (int c = lane; c < kN; c += 32) s += __expf(Sr[c] - m);
        #pragma unroll
        for (int off = 16; off > 0; off >>= 1) s += __shfl_xor(s, off, 32);
        if (lane == 0) { rowM[r] = m; rowR[r] = 1.0f / s; }
    }
    __syncthreads();

    // ---- Pass 3: normalize, stream attn to HBM, keep P in LDS ------------
    {
        float* attn_blk = OutAttn + ((size_t)bh * kN + q0) * kN;
        for (int idx = tid; idx < 16 * kN; idx += 128) {
            const int r = idx >> 11;            // kN = 2048
            const int c = idx & (kN - 1);
            const float p = __expf(sS[r * kSRow + c] - rowM[r]) * rowR[r];
            sS[r * kSRow + c] = p;
            attn_blk[(size_t)r * kN + c] = p;
        }
    }
    __syncthreads();

    // ---- Pass 4: O = P V (each wave owns a 16-wide d-slice) --------------
    const int dt = wave * 16;
    v8f o = {};
    for (int kk = 0; kk < kN / 32; ++kk) {
        const int kb2 = kk * 32;
        const float* Pr = sS + mA * kSRow + kb2;
        v16h pa, vb;
        #pragma unroll
        for (int i = 0; i < 8; ++i) {
            pa[i]     = (_Float16)Pr[     koffA + i];
            pa[8 + i] = (_Float16)Pr[ 8 + koffA + i];
        }
        #pragma unroll
        for (int i = 0; i < 16; ++i)
            vb[i] = (_Float16)Vb[(size_t)(kb2 + koffB + i) * kD + dt + nB];
        o = __builtin_amdgcn_wmma_f32_16x16x32_f16(false, pa, false, vb,
                                                   (short)0, o, false, false);
    }
    {
        float* orow = OutO + ((size_t)bh * kN + q0) * kD;
        #pragma unroll
        for (int j = 0; j < 8; ++j)
            orow[(size_t)(j + rbase) * kD + dt + nB] = o[j];
    }
}

extern "C" void kernel_launch(void* const* d_in, const int* in_sizes, int n_in,
                              void* d_out, int out_size, void* d_ws, size_t ws_size,
                              hipStream_t stream) {
    (void)in_sizes; (void)n_in; (void)out_size; (void)d_ws; (void)ws_size;
    const float* q    = (const float*)d_in[0];
    const float* k    = (const float*)d_in[1];
    const float* v    = (const float*)d_in[2];
    const int*   mask = (const int*)d_in[3];

    float* out_o    = (float*)d_out;                                   // (B,H,N,D)
    float* out_attn = out_o + (size_t)kB * kH * kN * kD;               // (B,H,N,N)

    const dim3 grid(kN / 16, kB * kH);   // 128 q-tiles x 32 (b,h)
    const dim3 block(128);
    const size_t lds_bytes = (size_t)16 * kSRow * sizeof(float)  // score strip
                           + 16 * sizeof(float)                  // rowM
                           + 16 * sizeof(float)                  // rowR
                           + 16 * sizeof(int);                   // mrow
    hipLaunchKernelGGL(ScaledDotProductAttention_30580167147801_kernel,
                       grid, block, lds_bytes, stream,
                       q, k, v, mask, out_o, out_attn);
}
